// FinalGen_15444702396815
// MI455X (gfx1250) — matrix operand order repaired and checked by reading, not compiled, not run on hardware
//
#include <hip/hip_runtime.h>
#include <hip/hip_bf16.h>
#include <stdint.h>

// CDNA5 / gfx1250 wave32 WMMA types
typedef __attribute__((ext_vector_type(16))) __bf16 v16bf;
typedef __attribute__((ext_vector_type(2)))  __bf16 v2bf;
typedef __attribute__((ext_vector_type(2)))  float  v2f;
typedef __attribute__((ext_vector_type(8)))  float  v8f;

#define DPAD 160   // 149 padded to 5 chunks of K=32
#define NH1  96
#define NH2  192

// Hardware fp32 -> bf16 conversion (lets clang pick v_cvt_*bf16* ops)
static __device__ __forceinline__ uint16_t f2bf(float f) {
  union { __bf16 b; uint16_t u; } v; v.b = (__bf16)f; return v.u;
}
static __device__ __forceinline__ uint32_t pack2bf(float lo, float hi) {
  v2f f; f.x = lo; f.y = hi;
  union { v2bf b; uint32_t u; } v;
  v.b = __builtin_convertvector(f, v2bf);
  return v.u;
}

union FragAB { v16bf v; uint32_t u[8]; uint16_t h[16]; };

static __device__ __forceinline__ v8f wmma_bf16(v16bf a, v16bf b, v8f c) {
  // D = A(16x32 bf16) x B(32x16 bf16) + C(16x16 f32)
  return __builtin_amdgcn_wmma_f32_16x16x32_bf16(false, a, false, b, (short)0, c,
                                                 false, false);
}

// A fragment (16x32 bf16, MxK) from a row-major bf16 LDS array.
// Lane layout: row = lane&15; lanes<16 hold K {0..7,16..23}, lanes>=16 hold {8..15,24..31}.
template <int STRIDE>
static __device__ __forceinline__ v16bf loadA(const uint16_t* base, int row, int kc, int lane) {
  FragAB f;
  const int kbase = kc * 32 + ((lane >> 4) << 3);
  const uint32_t* p = (const uint32_t*)(base + row * STRIDE + kbase);
#pragma unroll
  for (int j = 0; j < 4; ++j) f.u[j] = p[j];        // K = kbase + 0..7
#pragma unroll
  for (int j = 4; j < 8; ++j) f.u[j] = p[j + 4];    // K = kbase + 16..23
  return f.v;
}

// B fragment (32x16 bf16, KxN): lane holds column N=lane&15, 16 K values read contiguously from p.
static __device__ __forceinline__ v16bf loadB16(const uint16_t* p) {
  FragAB f;
  const uint32_t* q = (const uint32_t*)p;
#pragma unroll
  for (int i = 0; i < 8; ++i) f.u[i] = q[i];
  return f.v;
}

__global__ __launch_bounds__(32)
void fused_attn_mlp_gfx1250(const float* __restrict__ z_, const float* __restrict__ GSS,
                            const float* __restrict__ GSP,
                            const float* __restrict__ W1, const float* __restrict__ b1,
                            const float* __restrict__ W2, const float* __restrict__ b2,
                            float* __restrict__ out)
{
  __shared__ __align__(16) uint16_t Xs[32][DPAD];    // x, bf16, K-padded
  __shared__ __align__(16) uint16_t XTs[DPAD][32];   // x^T for attn @ x B operand
  __shared__ __align__(16) uint16_t As[32][32];      // attn (bf16)
  __shared__ __align__(16) uint16_t Ms[32][DPAD];    // m = attn @ x
  __shared__ __align__(16) uint16_t Hs[32][NH1];     // relu(m@W1+b1)
  __shared__ __align__(16) float    Gs[32][3];       // fp32 GSP rows for final add

  const int b  = blockIdx.x;
  const int l  = threadIdx.x;     // 0..31, one wave per batch
  const int lh = l >> 4;          // half-wave select
  const int ln = l & 15;

  // ---- stage 0: gather x = concat(z_,GSS,GSP), convert to bf16, build transpose ----
  {
    const float* zp = z_  + ((size_t)b * 32 + l) * 50;
    const float* sp = GSS + ((size_t)b * 32 + l) * 96;
    const float* pp = GSP + ((size_t)b * 32 + l) * 3;
    // z_ : 50 floats -> cols 0..49 (25 packed pairs into row-major Xs)
#pragma unroll 5
    for (int k = 0; k < 50; k += 2) {
      float a = zp[k], c = zp[k + 1];
      *(uint32_t*)&Xs[l][k] = pack2bf(a, c);
      XTs[k][l] = f2bf(a); XTs[k + 1][l] = f2bf(c);
    }
    // GSS : 96 floats -> cols 50..145
#pragma unroll 6
    for (int k = 0; k < 96; k += 2) {
      float a = sp[k], c = sp[k + 1];
      *(uint32_t*)&Xs[l][50 + k] = pack2bf(a, c);
      XTs[50 + k][l] = f2bf(a); XTs[51 + k][l] = f2bf(c);
    }
    // GSP : 3 floats -> cols 146..148 (+ fp32 copy for the final broadcast add)
#pragma unroll
    for (int k = 0; k < 3; ++k)  { float f = pp[k]; uint16_t h = f2bf(f);
                                   Xs[l][146 + k] = h; XTs[146 + k][l] = h; Gs[l][k] = f; }
    // zero pad cols/rows 149..159
#pragma unroll
    for (int k = 149; k < DPAD; ++k) { Xs[l][k] = 0; XTs[k][l] = 0; }
  }
  __syncthreads();

  // ---- stage 1: scores = x @ x^T / 8  (2x2 tiles, K = 5 chunks of 32) ----
  v8f S[2][2];
#pragma unroll
  for (int ti = 0; ti < 2; ++ti)
#pragma unroll
    for (int tj = 0; tj < 2; ++tj) S[ti][tj] = (v8f){0.f,0.f,0.f,0.f,0.f,0.f,0.f,0.f};

#pragma unroll
  for (int kc = 0; kc < 5; ++kc) {
    v16bf a0 = loadA<DPAD>(&Xs[0][0], ln,      kc, l);
    v16bf a1 = loadA<DPAD>(&Xs[0][0], 16 + ln, kc, l);
    v16bf bl = loadA<DPAD>(&Xs[0][0], ln,      kc, l);     // B cols = rows of x
    v16bf br = loadA<DPAD>(&Xs[0][0], 16 + ln, kc, l);
    S[0][0] = wmma_bf16(a0, bl, S[0][0]);
    S[0][1] = wmma_bf16(a0, br, S[0][1]);
    S[1][0] = wmma_bf16(a1, bl, S[1][0]);
    S[1][1] = wmma_bf16(a1, br, S[1][1]);
  }

  // ---- stage 2: row softmax in D-fragment layout (16-lane shfl_xor reductions) ----
#pragma unroll
  for (int ti = 0; ti < 2; ++ti) {
#pragma unroll
    for (int r = 0; r < 8; ++r) {
      float v0 = S[ti][0][r] * 0.125f;
      float v1 = S[ti][1][r] * 0.125f;
      float mx = fmaxf(v0, v1);
#pragma unroll
      for (int m = 1; m < 16; m <<= 1) mx = fmaxf(mx, __shfl_xor(mx, m, 32));
      float e0 = __expf(v0 - mx), e1 = __expf(v1 - mx);
      float s = e0 + e1;
#pragma unroll
      for (int m = 1; m < 16; m <<= 1) s += __shfl_xor(s, m, 32);
      float inv = 1.0f / s;
      const int R = 16 * ti + r + (lh << 3);
      As[R][ln]      = f2bf(e0 * inv);
      As[R][16 + ln] = f2bf(e1 * inv);
    }
  }
  __syncthreads();

  // ---- stage 3: m = attn @ x  (K = 32, one chunk; N = 10 tiles incl. zero pad) ----
  {
    v16bf aat0 = loadA<32>(&As[0][0], ln,      0, l);
    v16bf aat1 = loadA<32>(&As[0][0], 16 + ln, 0, l);
#pragma unroll
    for (int nj = 0; nj < 10; ++nj) {
      v16bf bx = loadA<32>(&XTs[0][0], nj * 16 + ln, 0, l);  // B col N=ln: feature over tokens
      v8f acc0 = (v8f){0.f,0.f,0.f,0.f,0.f,0.f,0.f,0.f};
      v8f acc1 = acc0;
      acc0 = wmma_bf16(aat0, bx, acc0);
      acc1 = wmma_bf16(aat1, bx, acc1);
      const int col = nj * 16 + ln;
#pragma unroll
      for (int r = 0; r < 8; ++r) {
        const int R0 = r + (lh << 3);
        Ms[R0][col]      = f2bf(acc0[r]);
        Ms[16 + R0][col] = f2bf(acc1[r]);
      }
    }
  }
  __syncthreads();

  // ---- stage 4: h1 = relu(m @ W1 + b1)  (K = 5 chunks, N = 6 tiles) ----
#pragma unroll 2
  for (int nj = 0; nj < 6; ++nj) {
    v8f acc0 = (v8f){0.f,0.f,0.f,0.f,0.f,0.f,0.f,0.f};
    v8f acc1 = acc0;
#pragma unroll
    for (int kc = 0; kc < 5; ++kc) {
      FragAB bw;
#pragma unroll
      for (int s = 0; s < 16; ++s) {
        const int krow = kc * 32 + (lh << 3) + (s & 7) + ((s >> 3) << 4);
        bw.h[s] = (krow < 149) ? f2bf(W1[(size_t)krow * NH1 + nj * 16 + ln]) : (uint16_t)0;
      }
      v16bf a0 = loadA<DPAD>(&Ms[0][0], ln,      kc, l);
      v16bf a1 = loadA<DPAD>(&Ms[0][0], 16 + ln, kc, l);
      acc0 = wmma_bf16(a0, bw.v, acc0);
      acc1 = wmma_bf16(a1, bw.v, acc1);
    }
    const int col  = nj * 16 + ln;
    const float bv = b1[col];
#pragma unroll
    for (int r = 0; r < 8; ++r) {
      const int R0 = r + (lh << 3);
      Hs[R0][col]      = f2bf(fmaxf(acc0[r] + bv, 0.0f));
      Hs[16 + R0][col] = f2bf(fmaxf(acc1[r] + bv, 0.0f));
    }
  }
  __syncthreads();

  // ---- stage 5: out = h1 @ W2 + b2 + GSP broadcast  (K = 3 chunks, N = 12 tiles) ----
  float* ob = out + (size_t)b * 32 * NH2;
#pragma unroll 2
  for (int nj = 0; nj < 12; ++nj) {
    v8f acc0 = (v8f){0.f,0.f,0.f,0.f,0.f,0.f,0.f,0.f};
    v8f acc1 = acc0;
#pragma unroll
    for (int kc = 0; kc < 3; ++kc) {
      FragAB bw;
#pragma unroll
      for (int s = 0; s < 16; ++s) {
        const int krow = kc * 32 + (lh << 3) + (s & 7) + ((s >> 3) << 4);   // < 96 always
        bw.h[s] = f2bf(W2[(size_t)krow * NH2 + nj * 16 + ln]);
      }
      v16bf a0 = loadA<NH1>(&Hs[0][0], ln,      kc, l);
      v16bf a1 = loadA<NH1>(&Hs[0][0], 16 + ln, kc, l);
      acc0 = wmma_bf16(a0, bw.v, acc0);
      acc1 = wmma_bf16(a1, bw.v, acc1);
    }
    const int col  = nj * 16 + ln;
    const float bv = b2[col];
    const int c3   = col % 3;
#pragma unroll
    for (int r = 0; r < 8; ++r) {
      const int R0 = r + (lh << 3);
      const int R1 = 16 + R0;
      ob[(size_t)R0 * NH2 + col] = acc0[r] + bv + Gs[R0][c3];
      ob[(size_t)R1 * NH2 + col] = acc1[r] + bv + Gs[R1][c3];
    }
  }
}

extern "C" void kernel_launch(void* const* d_in, const int* in_sizes, int n_in,
                              void* d_out, int out_size, void* d_ws, size_t ws_size,
                              hipStream_t stream) {
  (void)n_in; (void)out_size; (void)d_ws; (void)ws_size;
  const float* z_  = (const float*)d_in[0];
  const float* GSS = (const float*)d_in[1];
  const float* GSP = (const float*)d_in[2];
  const float* W1  = (const float*)d_in[3];
  const float* b1  = (const float*)d_in[4];
  const float* W2  = (const float*)d_in[5];
  const float* b2  = (const float*)d_in[6];
  const int B = in_sizes[0] / (32 * 50);   // 8192 batches
  fused_attn_mlp_gfx1250<<<B, 32, 0, stream>>>(z_, GSS, GSP, W1, b1, W2, b2,
                                               (float*)d_out);
}
